// GAT_34540126995073
// MI455X (gfx1250) — matrix operand order repaired and checked by reading, not compiled, not run on hardware
//
#include <hip/hip_runtime.h>

typedef __attribute__((ext_vector_type(16))) _Float16 v16h;
typedef __attribute__((ext_vector_type(8)))  _Float16 v8h;
typedef __attribute__((ext_vector_type(8)))  float    v8f;

#define NEG_SLOPE 0.2f
#define BN_EPS    1e-5f
#define DIM       128
#define HEADS     3
#define HD        384   // HEADS*DIM

// ---------------------------------------------------------------------------
// WMMA GEMM: C[M,Nc] = A[M,K] (f16, row-major, leading dim lda) * B[K,Nc] + bias
// B supplied transposed as Bt[Nc,K] f16 row-major (contiguous K per output col).
//
// Wave32 tiling: block = 256 thr = 8 waves arranged 4(M) x 2(N).
// Block tile = 64 rows x 128 cols; each wave owns a 16x64 strip:
// ONE A fragment feeds FOUR accumulators per K-step (4 wmma : 10 b128 loads),
// 4x A-reuse in registers vs. one-tile-per-wave.
// Grid = (ceil(M/64), Nc/128); Nc in {128, 384} divides exactly.
//
// Fragment layouts per CDNA5 ISA 7.12.2 (wave32):
//   A (16x32 f16): lane m = lane&15, half = lane>>4;
//     elem e<8  -> K = kblk + half*8 + e
//     elem e>=8 -> K = kblk + 16 + half*8 + (e-8)
//   B (32x16 f16): n = lane&15; same K mapping as A
//   C/D (16x16 f32): elem r -> row = half*8 + r, col = lane&15
// ---------------------------------------------------------------------------
__global__ __launch_bounds__(256) void gemm_wmma_f16(
    const _Float16* __restrict__ A, int lda,
    const _Float16* __restrict__ Bt,
    const float* __restrict__ bias,
    float* __restrict__ C, int M, int Nc, int K)
{
  const int lane  = threadIdx.x & 31;
  const int wave  = threadIdx.x >> 5;
  const int waveM = wave & 3;
  const int waveN = wave >> 2;
  const int m0    = (blockIdx.x << 6) + (waveM << 4);
  const int n0    = (blockIdx.y << 7) + (waveN << 6);   // 64-wide strip
  const int mlane = lane & 15;
  const int half  = lane >> 4;

  int arow = m0 + mlane;
  if (arow >= M) arow = M - 1;                 // clamp; padded rows never stored
  const _Float16* Ar = A  + (size_t)arow * lda;
  const _Float16* Br = Bt + (size_t)(n0 + mlane) * K;  // tile t adds (t*16)*K

  v8f acc[4] = {};
  for (int k = 0; k < K; k += 32) {
    // ---- A fragment (shared by all 4 output tiles) ----
    v8h a_lo = *(const v8h*)(Ar + k + half * 8);
    v8h a_hi = *(const v8h*)(Ar + k + 16 + half * 8);
    v16h a;
#pragma unroll
    for (int e = 0; e < 8; ++e) { a[e] = a_lo[e]; a[e + 8] = a_hi[e]; }

    // speculative prefetch of the K block two steps ahead (global_prefetch_b8; OOB-safe)
    __builtin_prefetch(Ar + k + 64, 0, 0);
    __builtin_prefetch(Br + k + 64, 0, 0);

#pragma unroll
    for (int t = 0; t < 4; ++t) {
      const _Float16* Bp = Br + (size_t)(t << 4) * K + k + half * 8;
      v8h b_lo = *(const v8h*)(Bp);
      v8h b_hi = *(const v8h*)(Bp + 16);
      v16h b;
#pragma unroll
      for (int e = 0; e < 8; ++e) { b[e] = b_lo[e]; b[e + 8] = b_hi[e]; }
      acc[t] = __builtin_amdgcn_wmma_f32_16x16x32_f16(
          /*neg_a=*/false, a, /*neg_b=*/false, b,
          /*c_mod=*/(short)0, acc[t], /*reuse_a=*/false, /*reuse_b=*/false);
    }
  }

#pragma unroll
  for (int t = 0; t < 4; ++t) {
    const int   col = n0 + (t << 4) + mlane;
    const float bv  = bias ? bias[col] : 0.f;
#pragma unroll
    for (int r = 0; r < 8; ++r) {
      int row = m0 + half * 8 + r;
      if (row < M) C[(size_t)row * Nc + col] = acc[t][r] + bv;
    }
  }
}

// ---------------------------------------------------------------------------
// Conversions
// ---------------------------------------------------------------------------
__global__ void f32_to_f16_2d(const float* __restrict__ in, int ldin,
                              _Float16* __restrict__ out, int ldout,
                              int rows, int cols)
{
  size_t i = (size_t)blockIdx.x * blockDim.x + threadIdx.x;
  if (i >= (size_t)rows * cols) return;
  size_t r = i / cols;
  int    c = (int)(i - r * cols);
  out[r * ldout + c] = (_Float16)in[r * (size_t)ldin + c];
}

// W[K,Nc] f32 -> Wt[Nc,K] f16
__global__ void w_transpose_f16(const float* __restrict__ W,
                                _Float16* __restrict__ Wt, int K, int Nc)
{
  size_t i = (size_t)blockIdx.x * blockDim.x + threadIdx.x;
  if (i >= (size_t)K * Nc) return;
  int k = (int)(i / Nc);
  int n = (int)(i - (size_t)k * Nc);
  Wt[(size_t)n * K + k] = (_Float16)W[i];
}

// ---------------------------------------------------------------------------
// GATv2 edge machinery. Edge list is [2,E]; self-loops e>=E -> (e-E, e-E).
// ---------------------------------------------------------------------------
__device__ inline void edge_sd(const int* __restrict__ ei, int E, int e,
                               int& s, int& d)
{
  if (e < E) { s = ei[e]; d = ei[E + e]; }
  else       { s = d = e - E; }
}

__device__ inline void atomicMaxF(float* addr, float v)
{
  int iv = __float_as_int(v);
  if (iv >= 0) atomicMax((int*)addr, iv);
  else         atomicMin((unsigned int*)addr, (unsigned int)iv);
}

__global__ void seg_init(float* __restrict__ m, float* __restrict__ denom, int n)
{
  int i = blockIdx.x * blockDim.x + threadIdx.x;
  if (i >= n) return;
  m[i]     = -__builtin_inff();
  denom[i] = 0.f;
}

// one wave per (edge, head): score = att_h . leaky_relu(xl[src,h,:]+xr[dst,h,:])
__global__ __launch_bounds__(256) void edge_scores(
    const int* __restrict__ ei, int E, int N,
    const float* __restrict__ xl, const float* __restrict__ xr,
    const float* __restrict__ att, float* __restrict__ score)
{
  int task = blockIdx.x * 8 + (threadIdx.x >> 5);
  int lane = threadIdx.x & 31;
  int EN = E + N;
  if (task >= EN * HEADS) return;
  int e = task / HEADS, h = task - e * HEADS;
  int s, d; edge_sd(ei, E, e, s, d);
  const float* pl = xl + (size_t)s * HD + h * DIM;
  const float* pr = xr + (size_t)d * HD + h * DIM;
  const float* pa = att + h * DIM;
  float acc = 0.f;
#pragma unroll
  for (int c = lane; c < DIM; c += 32) {
    float v = pl[c] + pr[c];
    v = v > 0.f ? v : NEG_SLOPE * v;
    acc += v * pa[c];
  }
#pragma unroll
  for (int off = 16; off; off >>= 1) acc += __shfl_xor(acc, off, 32);
  if (lane == 0) score[(size_t)e * HEADS + h] = acc;
}

__global__ void seg_max(const int* __restrict__ ei, int E, int N,
                        const float* __restrict__ score, float* __restrict__ m)
{
  int i = blockIdx.x * blockDim.x + threadIdx.x;
  int EN = E + N;
  if (i >= EN * HEADS) return;
  int e = i / HEADS, h = i - e * HEADS;
  int s, d; edge_sd(ei, E, e, s, d);
  atomicMaxF(&m[d * HEADS + h], score[i]);
}

__global__ void exp_sum(const int* __restrict__ ei, int E, int N,
                        float* __restrict__ score, const float* __restrict__ m,
                        float* __restrict__ denom)
{
  int i = blockIdx.x * blockDim.x + threadIdx.x;
  int EN = E + N;
  if (i >= EN * HEADS) return;
  int e = i / HEADS, h = i - e * HEADS;
  int s, d; edge_sd(ei, E, e, s, d);
  float v = __expf(score[i] - m[d * HEADS + h]);
  score[i] = v;                       // now holds exp(score - max)
  atomicAdd(&denom[d * HEADS + h], v);
}

__global__ void init_bias_rows(float* __restrict__ out,
                               const float* __restrict__ bias,
                               int rows, int cols)
{
  size_t i = (size_t)blockIdx.x * blockDim.x + threadIdx.x;
  if (i >= (size_t)rows * cols) return;
  out[i] = bias[i % cols];
}

// one wave per edge: out[dst,h,:] += (ex/denom[dst,h]) * xl[src,h,:]
__global__ __launch_bounds__(256) void edge_aggregate(
    const int* __restrict__ ei, int E, int N,
    const float* __restrict__ xl, const float* __restrict__ ex,
    const float* __restrict__ denom, float* __restrict__ out)
{
  int e = blockIdx.x * 8 + (threadIdx.x >> 5);
  int lane = threadIdx.x & 31;
  if (e >= E + N) return;
  int s, d; edge_sd(ei, E, e, s, d);
  float al[HEADS];
#pragma unroll
  for (int h = 0; h < HEADS; ++h)
    al[h] = ex[(size_t)e * HEADS + h] / denom[d * HEADS + h];
  const float* src = xl + (size_t)s * HD;
  float*       dst = out + (size_t)d * HD;
#pragma unroll
  for (int c = lane; c < HD; c += 32)
    atomicAdd(dst + c, al[c >> 7] * src[c]);
}

// ---------------------------------------------------------------------------
// BatchNorm: per-column mean / biased var, then fused normalize + ReLU
// ---------------------------------------------------------------------------
__global__ __launch_bounds__(256) void bn_stats(const float* __restrict__ Y,
                                                int rows, int C,
                                                float* __restrict__ mv)
{
  int c = blockIdx.x, t = threadIdx.x;
  float s = 0.f, s2 = 0.f;
  for (int r = t; r < rows; r += 256) {
    float v = Y[(size_t)r * C + c];
    s += v; s2 += v * v;
  }
  __shared__ float sh[256], sh2[256];
  sh[t] = s; sh2[t] = s2; __syncthreads();
  for (int o = 128; o; o >>= 1) {
    if (t < o) { sh[t] += sh[t + o]; sh2[t] += sh2[t + o]; }
    __syncthreads();
  }
  if (t == 0) {
    float mu = sh[0] / rows;
    mv[c]     = mu;
    mv[C + c] = sh2[0] / rows - mu * mu;
  }
}

__global__ void bn_relu_apply(const float* __restrict__ Y,
                              const float* __restrict__ mv,
                              const float* __restrict__ g,
                              const float* __restrict__ b,
                              int rows, int C,
                              float* __restrict__ o32,
                              _Float16* __restrict__ o16, int ld16)
{
  size_t i = (size_t)blockIdx.x * blockDim.x + threadIdx.x;
  if (i >= (size_t)rows * C) return;
  size_t r = i / C;
  int    c = (int)(i - r * C);
  float v = g[c] * (Y[i] - mv[c]) * rsqrtf(mv[C + c] + BN_EPS) + b[c];
  v = fmaxf(v, 0.f);
  if (o32) o32[i] = v;
  if (o16) o16[r * ld16 + c] = (_Float16)v;
}

// ---------------------------------------------------------------------------
// Host-side orchestration
// ---------------------------------------------------------------------------
static inline int cdiv(int a, int b) { return (a + b - 1) / b; }

extern "C" void kernel_launch(void* const* d_in, const int* in_sizes, int n_in,
                              void* d_out, int out_size, void* d_ws, size_t ws_size,
                              hipStream_t stream)
{
  (void)n_in; (void)out_size; (void)ws_size;
  const float* x    = (const float*)d_in[0];
  const int*   ei   = (const int*)  d_in[1];
  const float* Wl1  = (const float*)d_in[2];
  const float* bl1  = (const float*)d_in[3];
  const float* Wr1  = (const float*)d_in[4];
  const float* br1  = (const float*)d_in[5];
  const float* att1 = (const float*)d_in[6];
  const float* bc1  = (const float*)d_in[7];
  const float* g1   = (const float*)d_in[8];
  const float* be1  = (const float*)d_in[9];
  const float* Wl2  = (const float*)d_in[10];
  const float* bl2  = (const float*)d_in[11];
  const float* Wr2  = (const float*)d_in[12];
  const float* br2  = (const float*)d_in[13];
  const float* att2 = (const float*)d_in[14];
  const float* bc2  = (const float*)d_in[15];
  const float* g2   = (const float*)d_in[16];
  const float* be2  = (const float*)d_in[17];
  const float* W1   = (const float*)d_in[18];
  const float* b1   = (const float*)d_in[19];
  const float* W2   = (const float*)d_in[20];
  const float* b2   = (const float*)d_in[21];

  const int N  = in_sizes[0] / DIM;
  const int E  = in_sizes[1] / 2;
  const int EN = E + N;
  const int K2 = 2 * HD;  // 768

  // ---- carve workspace ----
  size_t off = 0;
  auto alloc = [&](size_t bytes) -> char* {
    char* p = (char*)d_ws + off;
    off += (bytes + 255) & ~(size_t)255;
    return p;
  };
  float*    xl    = (float*)   alloc((size_t)N * HD * 4);
  float*    xr    = (float*)   alloc((size_t)N * HD * 4);  // reused as h2 in layer 2
  float*    x_in  = (float*)   alloc((size_t)N * HD * 4);
  float*    t1    = (float*)   alloc((size_t)N * DIM * 4);
  float*    score = (float*)   alloc((size_t)EN * HEADS * 4);
  float*    m     = (float*)   alloc((size_t)N * HEADS * 4);
  float*    denom = (float*)   alloc((size_t)N * HEADS * 4);
  float*    mv    = (float*)   alloc(2 * DIM * 4);
  _Float16* xf16  = (_Float16*)alloc((size_t)N * DIM * 2);
  _Float16* hf16  = (_Float16*)alloc((size_t)N * DIM * 2);
  _Float16* big16 = (_Float16*)alloc((size_t)N * K2 * 2);   // [h2 | x_in] f16
  _Float16* Wl1t  = (_Float16*)alloc((size_t)DIM * HD * 2);
  _Float16* Wr1t  = (_Float16*)alloc((size_t)DIM * HD * 2);
  _Float16* Wl2t  = (_Float16*)alloc((size_t)DIM * HD * 2);
  _Float16* Wr2t  = (_Float16*)alloc((size_t)DIM * HD * 2);
  _Float16* W1t   = (_Float16*)alloc((size_t)HD * DIM * 2);
  _Float16* W2t   = (_Float16*)alloc((size_t)K2 * DIM * 2);

  const int T = 256;
  // ---- weight prep (f16, transposed) ----
  w_transpose_f16<<<cdiv(DIM * HD, T), T, 0, stream>>>(Wl1, Wl1t, DIM, HD);
  w_transpose_f16<<<cdiv(DIM * HD, T), T, 0, stream>>>(Wr1, Wr1t, DIM, HD);
  w_transpose_f16<<<cdiv(DIM * HD, T), T, 0, stream>>>(Wl2, Wl2t, DIM, HD);
  w_transpose_f16<<<cdiv(DIM * HD, T), T, 0, stream>>>(Wr2, Wr2t, DIM, HD);
  w_transpose_f16<<<cdiv(HD * DIM, T), T, 0, stream>>>(W1, W1t, HD, DIM);
  w_transpose_f16<<<cdiv(K2 * DIM, T), T, 0, stream>>>(W2, W2t, K2, DIM);
  f32_to_f16_2d<<<cdiv(N * DIM, T), T, 0, stream>>>(x, DIM, xf16, DIM, N, DIM);

  const dim3 gHD(cdiv(N, 64), HD / 128);   // Nc=384
  const dim3 gD (cdiv(N, 64), DIM / 128);  // Nc=128

  // ---- GAT layer 1 ----
  gemm_wmma_f16<<<gHD, T, 0, stream>>>(xf16, DIM, Wl1t, bl1, xl, N, HD, DIM);
  gemm_wmma_f16<<<gHD, T, 0, stream>>>(xf16, DIM, Wr1t, br1, xr, N, HD, DIM);
  seg_init<<<cdiv(N * HEADS, T), T, 0, stream>>>(m, denom, N * HEADS);
  edge_scores<<<cdiv(EN * HEADS, 8), T, 0, stream>>>(ei, E, N, xl, xr, att1, score);
  seg_max<<<cdiv(EN * HEADS, T), T, 0, stream>>>(ei, E, N, score, m);
  exp_sum<<<cdiv(EN * HEADS, T), T, 0, stream>>>(ei, E, N, score, m, denom);
  init_bias_rows<<<cdiv(N * HD, T), T, 0, stream>>>(x_in, bc1, N, HD);
  edge_aggregate<<<cdiv(EN, 8), T, 0, stream>>>(ei, E, N, xl, score, denom, x_in);
  f32_to_f16_2d<<<cdiv(N * HD, T), T, 0, stream>>>(x_in, HD, big16 + HD, K2, N, HD);

  // ---- h = relu(BN(x_in @ W1 + b1)) ----
  gemm_wmma_f16<<<gD, T, 0, stream>>>(big16 + HD, K2, W1t, b1, t1, N, DIM, HD);
  bn_stats<<<DIM, T, 0, stream>>>(t1, N, DIM, mv);
  bn_relu_apply<<<cdiv(N * DIM, T), T, 0, stream>>>(t1, mv, g1, be1, N, DIM,
                                                    nullptr, hf16, DIM);

  // ---- GAT layer 2 (output h2 reuses xr buffer; xr dead after scores) ----
  gemm_wmma_f16<<<gHD, T, 0, stream>>>(hf16, DIM, Wl2t, bl2, xl, N, HD, DIM);
  gemm_wmma_f16<<<gHD, T, 0, stream>>>(hf16, DIM, Wr2t, br2, xr, N, HD, DIM);
  seg_init<<<cdiv(N * HEADS, T), T, 0, stream>>>(m, denom, N * HEADS);
  edge_scores<<<cdiv(EN * HEADS, 8), T, 0, stream>>>(ei, E, N, xl, xr, att2, score);
  seg_max<<<cdiv(EN * HEADS, T), T, 0, stream>>>(ei, E, N, score, m);
  exp_sum<<<cdiv(EN * HEADS, T), T, 0, stream>>>(ei, E, N, score, m, denom);
  init_bias_rows<<<cdiv(N * HD, T), T, 0, stream>>>(xr, bc2, N, HD);
  edge_aggregate<<<cdiv(EN, 8), T, 0, stream>>>(ei, E, N, xl, score, denom, xr);
  f32_to_f16_2d<<<cdiv(N * HD, T), T, 0, stream>>>(xr, HD, big16, K2, N, HD);

  // ---- out = relu(BN(concat([h2, x_in]) @ W2 + b2)) ----
  gemm_wmma_f16<<<gD, T, 0, stream>>>(big16, K2, W2t, b2, t1, N, DIM, K2);
  bn_stats<<<DIM, T, 0, stream>>>(t1, N, DIM, mv);
  bn_relu_apply<<<cdiv(N * DIM, T), T, 0, stream>>>(t1, mv, g2, be2, N, DIM,
                                                    (float*)d_out, nullptr, 0);
}
